// ASK_cuda_55241869361274
// MI455X (gfx1250) — matrix operand order repaired and checked by reading, not compiled, not run on hardware
//
#include <hip/hip_runtime.h>

typedef __attribute__((ext_vector_type(16))) _Float16 v16h;
typedef __attribute__((ext_vector_type(8)))  float    v8f;

#define INP   128
#define OCD   32
#define HH    56
#define WW    56
#define BB    32
#define KTOT  512          // 128 ic * 4 taps
#define NKK   16           // K steps of 32
#define LDSC  64           // padded cols in LDS (slots 0..57 valid)
#define LDSROWS 3

// ---------------------------------------------------------------------------
// Kernel 1: pre-swizzle weights fp32 -> f16 into CDNA5 A-fragment layout.
// A frag (16-bit A 16x32), contiguous K order used here: per lane 16 halves;
//   lanes 0-15:  K=j
//   lanes 16-31: K=16+j
//   M = lane & 15
// Storage: Abuf[((g*2+mt)*16 + kk)*32*16 + lane*16 + j]
// ---------------------------------------------------------------------------
__global__ void ask_prep_wfrag(const float* __restrict__ w, _Float16* __restrict__ abuf) {
    int hidx = blockIdx.x * 256 + threadIdx.x;        // 0 .. 65535
    int j    = hidx & 15;
    int lane = (hidx >> 4) & 31;
    int kk   = (hidx >> 9) & 15;
    int mt   = (hidx >> 13) & 1;
    int g    = (hidx >> 14) & 3;

    int lo   = (lane < 16);
    int koff = (lo ? 0 : 16) + j;
    int k    = kk * 32 + koff;                        // k = ic*4 + t (matches ref cols)
    int oc   = mt * 16 + (lane & 15);
    float v  = w[oc * 2048 + g * KTOT + k];
    abuf[hidx] = (_Float16)v;
}

// ---------------------------------------------------------------------------
// Kernel 2: implicit-GEMM sparse-tap conv via v_wmma_f32_16x16x32_f16.
// One block per (b,h): stage x[b, :, h-1..h+1, -1..56] as f16 in LDS,
// 8 waves = (4 groups) x (2 N-halves); each wave: 2 N-subtiles x 16 K-steps
// x 2 M-tiles -> 64 WMMAs.
// ---------------------------------------------------------------------------
__global__ __launch_bounds__(256) void ask_conv_wmma(
        const float* __restrict__ x,
        const _Float16* __restrict__ abuf,
        float* __restrict__ out) {

    __shared__ _Float16 lds_x[INP * LDSROWS * LDSC];  // 48 KB

    const int h  = blockIdx.x;
    const int bb = blockIdx.y;
    const int tid = threadIdx.x;

    // ---- stage x slab (fp32 global -> f16 LDS), zero-padded halo ----
    for (int e = tid; e < INP * LDSROWS * LDSC; e += 256) {
        int c   = e & (LDSC - 1);
        int rr  = e >> 6;            // ic*3 + r
        int r   = rr % 3;
        int ic  = rr / 3;
        int hr  = h - 1 + r;
        int wc  = c - 1;
        float v = 0.0f;
        if (hr >= 0 && hr < HH && c >= 1 && c < (WW + 1)) {
            v = x[((bb * INP + ic) * HH + hr) * WW + wc];
        }
        lds_x[e] = (_Float16)v;
    }
    __syncthreads();

    const int wv   = tid >> 5;
    const int lane = tid & 31;
    const int g    = wv >> 1;        // pattern group 0..3
    const int nh   = wv & 1;         // N-half
    const int col  = lane & 15;      // N within tile / M within A
    const int hi   = lane >> 4;      // lane half -> K half
    const int drg  = g >> 1;         // tap row base
    const int dcg  = g & 1;          // tap col base

    const size_t aStride = 32 * 16;                          // halves per (g,mt,kk)
    const size_t aOff0 = (size_t)((g * 2 + 0) * 16) * aStride + (size_t)lane * 16;
    const size_t aOff1 = (size_t)((g * 2 + 1) * 16) * aStride + (size_t)lane * 16;

    for (int s = 0; s < 2; ++s) {
        const int n0 = (nh * 2 + s) * 16;
        const int n  = n0 + col;                             // w position (slot base)
        v8f c0 = {};
        v8f c1 = {};

        for (int kk = 0; kk < NKK; ++kk) {
            // A fragments: contiguous 32B per lane -> 2x global_load_b128 each
            v16h a0 = *(const v16h*)(abuf + aOff0 + (size_t)kk * aStride);
            v16h a1 = *(const v16h*)(abuf + aOff1 + (size_t)kk * aStride);
            if (kk + 1 < NKK) {
                __builtin_prefetch(abuf + aOff0 + (size_t)(kk + 1) * aStride, 0, 1);
            }

            // B fragment gather from LDS.
            // B 32x16 (16-bit), order used here: lanes 0-15 hold K=0..15, lanes 16-31 hold K=16..31,
            // half j of v16h <-> K = hi*16 + j.  k = kk*32 + hi*16 + jc*4 + t
            //  -> ic = kk*8 + hi*4 + jc, tap index t -> (dr,dc).
            const int icb = kk * 8 + hi * 4;
            v16h bf;
#pragma unroll
            for (int jc = 0; jc < 4; ++jc) {
                const _Float16* rowp = &lds_x[(icb + jc) * (LDSROWS * LDSC) + n];
#pragma unroll
                for (int t = 0; t < 4; ++t) {
                    bf[jc * 4 + t] = rowp[(drg + (t >> 1)) * LDSC + dcg + (t & 1)];
                }
            }

            c0 = __builtin_amdgcn_wmma_f32_16x16x32_f16(
                     false, a0, false, bf, (short)0, c0, false, false);
            c1 = __builtin_amdgcn_wmma_f32_16x16x32_f16(
                     false, a1, false, bf, (short)0, c1, false, false);
        }

        // ---- store: C/D 16x16 f32 layout: VGPR r -> M = hi*8 + r, N = col ----
        const int w = n0 + col;
        if (w < WW) {
#pragma unroll
            for (int r = 0; r < 8; ++r) {
                const int m   = hi * 8 + r;
                const int oc0 = g * OCD + m;             // mt=0 channels
                out[((bb * 128 + oc0) * HH + h) * WW + w]      = c0[r];
                out[((bb * 128 + oc0 + 16) * HH + h) * WW + w] = c1[r];
            }
        }
    }
}

// ---------------------------------------------------------------------------
extern "C" void kernel_launch(void* const* d_in, const int* in_sizes, int n_in,
                              void* d_out, int out_size, void* d_ws, size_t ws_size,
                              hipStream_t stream) {
    const float* x = (const float*)d_in[0];   // (32,128,56,56) fp32
    const float* w = (const float*)d_in[1];   // (32,2048) fp32
    float* out = (float*)d_out;               // (32,128,56,56) fp32
    _Float16* abuf = (_Float16*)d_ws;         // 65536 halves = 128 KB

    // 1) swizzle weights into WMMA A-fragment layout (f16)
    ask_prep_wfrag<<<256, 256, 0, stream>>>(w, abuf);

    // 2) implicit-GEMM conv, one block per (h, b)
    dim3 grid(HH, BB);
    ask_conv_wmma<<<grid, 256, 0, stream>>>(x, abuf, out);
}
